// ModernFinGATWithSector_36043365548676
// MI455X (gfx1250) — matrix operand order repaired and checked by reading, not compiled, not run on hardware
//
#include <hip/hip_runtime.h>
#include <math.h>

// ---------------- types ----------------
typedef __bf16 bf16_t;
typedef __attribute__((ext_vector_type(16))) __bf16 v16bf;
typedef __attribute__((ext_vector_type(8)))  __bf16 v8bf;
typedef __attribute__((ext_vector_type(8)))  float  v8f;

#define ENC_NEGINF 0x007FFFFFu   // fenc(-inf)

__device__ inline bf16_t f2bf(float f) {
  unsigned u = __float_as_uint(f);
  u += 0x7FFFu + ((u >> 16) & 1u);              // round-to-nearest-even
  unsigned short s = (unsigned short)(u >> 16);
  return __builtin_bit_cast(bf16_t, s);
}

// order-preserving float<->uint encoding for atomic max on floats
__device__ inline unsigned fenc(float f) {
  unsigned u = __float_as_uint(f);
  return (u & 0x80000000u) ? ~u : (u | 0x80000000u);
}
__device__ inline float fdec(unsigned u) {
  unsigned v = (u & 0x80000000u) ? (u & 0x7FFFFFFFu) : ~u;
  return __uint_as_float(v);
}

__device__ inline void edge_sd(const int* ei, int E, int e, int& s, int& d) {
  if (e < E) { s = ei[e]; d = ei[E + e]; }
  else       { s = e - E; d = e - E; }          // appended self-loops
}

// ---------------- B-matrix pre-pack ----------------
// Pack fp32 weight into bf16 tile-swizzled layout [K/32][Ncol/64][2048] where a
// 2048-element block is fragment-ready for wave32 WMMA:
//   inner = colTile(0..3)*512 + lane(0..31)*16 + e(0..15)
//   col   = nb*64 + colTile*16 + (lane&15)
//   k     = kb*32 + (lane>>4)*16 + e
// trans=1 reads W[col*ldw + k] (i.e. B = W^T), trans=0 reads W[k*ldw + col].
__global__ void pack_b(const float* __restrict__ W, int ldw, int trans,
                       bf16_t* __restrict__ dst, int K, int Ncol) {
  long idx = (long)blockIdx.x * blockDim.x + threadIdx.x;
  if (idx >= (long)K * Ncol) return;
  int NB = Ncol >> 6;
  long blk = idx >> 11;
  int inner = (int)(idx & 2047);
  int kb = (int)(blk / NB), nb = (int)(blk % NB);
  int colTile = inner >> 9;
  int lane = (inner >> 4) & 31;
  int e = inner & 15;
  int col = nb * 64 + colTile * 16 + (lane & 15);
  int k = kb * 32 + (lane >> 4) * 16 + e;
  float v = trans ? W[(long)col * ldw + k] : W[(long)k * ldw + col];
  dst[idx] = f2bf(v);
}

// ---------------- GEMM helpers ----------------
// Stage one 64x32 A tile (fp32 -> bf16, fragment-ready swizzle).
__device__ inline void stage_a_tile(const float* arow, int k0, bool inb,
                                    bf16_t* dst, int rt, int rr, int eo) {
  float va[16];
  if (inb) {
    const float4* p = (const float4*)(arow + k0);
    float4 a0 = p[0], a1 = p[1], a2 = p[2], a3 = p[3];
    va[0]=a0.x; va[1]=a0.y; va[2]=a0.z; va[3]=a0.w;
    va[4]=a1.x; va[5]=a1.y; va[6]=a1.z; va[7]=a1.w;
    va[8]=a2.x; va[9]=a2.y; va[10]=a2.z; va[11]=a2.w;
    va[12]=a3.x; va[13]=a3.y; va[14]=a3.z; va[15]=a3.w;
  } else {
#pragma unroll
    for (int i = 0; i < 16; i++) va[i] = 0.0f;
  }
  v8bf lo, hi;
#pragma unroll
  for (int i = 0; i < 8; i++) { lo[i] = f2bf(va[i]); hi[i] = f2bf(va[8 + i]); }
  *(v8bf*)(dst + rt * 512 + (rr + 2 * eo) * 16)     = lo;   // k = ch+0..7  -> lane rr+ch, e 0..7
  *(v8bf*)(dst + rt * 512 + (rr + 2 * eo) * 16 + 8) = hi;   // k = ch+8..15 -> lane rr+ch, e 8..15
}

// Async-DMA one pre-packed 2048-element bf16 B tile into LDS (ASYNCcnt-tracked).
// Each of the 128 threads copies two 16-byte chunks.
__device__ inline void async_copy_b(const bf16_t* src, bf16_t* dstLds, int tid) {
  unsigned d0 = (unsigned)(size_t)(dstLds + tid * 8);          // lane LDS byte addr
  unsigned d1 = (unsigned)(size_t)(dstLds + 1024 + tid * 8);
  const bf16_t* s0 = src + tid * 8;
  const bf16_t* s1 = src + 1024 + tid * 8;
  asm volatile("global_load_async_to_lds_b128 %0, %1, off\n\t"
               "global_load_async_to_lds_b128 %2, %3, off"
               :: "v"(d0), "v"(s0), "v"(d1), "v"(s1)
               : "memory");
}

__device__ inline void wait_async0() {
  asm volatile("s_wait_asynccnt 0" ::: "memory");
}

// ---------------- WMMA GEMM ----------------
// C[M,Ncol] = act( A[M,K](fp32) @ Bpacked(bf16) + bias + (Cin) )
// Requires K % 32 == 0, Ncol % 64 == 0 (true at all call sites).
// Block = 128 threads (4 waves); block tile 64x64; wave tile 32x32 (2x2 WMMA).
// Double-buffered LDS: async B DMA + A staging for tile kb+1 overlap WMMAs on kb.
__global__ __launch_bounds__(128)
void gemm_wmma(const float* __restrict__ A, int lda,
               const bf16_t* __restrict__ Bp,
               const float* __restrict__ bias,
               const float* __restrict__ Cin,
               float* __restrict__ C, int ldc,
               int M, int Ncol, int K, int act) {
  __shared__ __align__(32) bf16_t lA[2][2048];
  __shared__ __align__(32) bf16_t lB[2][2048];

  int tid  = threadIdx.x;
  int wave = tid >> 5;
  int lane = tid & 31;
  int half = lane >> 4;
  int l16  = lane & 15;
  int waveRow = wave >> 1;      // 0..1
  int waveCol = wave & 1;       // 0..1
  int blockM = blockIdx.y * 64;
  int blockN = blockIdx.x * 64;
  int NB = Ncol >> 6;

  v8f acc[2][2];
#pragma unroll
  for (int i = 0; i < 2; i++)
#pragma unroll
    for (int j = 0; j < 2; j++)
#pragma unroll
      for (int q = 0; q < 8; q++) acc[i][j][q] = 0.0f;

  // A staging assignment: thread owns row r = tid>>1, K-half ch = (tid&1)*16
  int r  = tid >> 1;
  int ch = (tid & 1) << 4;
  int rt = r >> 4, rr = r & 15;
  int eo = ch >> 1;             // 0 or 8 : 2*eo (= ch) is the lane offset, 0 or 16
  int gr = blockM + r;
  bool inb = (gr < M);
  const float* arow = A + (long)gr * lda + ch;

  int nk = K >> 5;

  // ---- prologue: stage tile 0 ----
  stage_a_tile(arow, 0, inb, lA[0], rt, rr, eo);
  async_copy_b(Bp + (long)blockIdx.x * 2048, lB[0], tid);
  wait_async0();
  __syncthreads();

  for (int kb = 0; kb < nk; kb++) {
    int p = kb & 1;
    if (kb + 1 < nk) {
      stage_a_tile(arow, (kb + 1) << 5, inb, lA[p ^ 1], rt, rr, eo);
      async_copy_b(Bp + ((long)(kb + 1) * NB + blockIdx.x) * 2048, lB[p ^ 1], tid);
    }

    // ---- fragments: one contiguous 32B run per lane ----
    v16bf afrag[2], bfrag[2];
    afrag[0] = *(const v16bf*)(lA[p] + (waveRow * 2 + 0) * 512 + lane * 16);
    afrag[1] = *(const v16bf*)(lA[p] + (waveRow * 2 + 1) * 512 + lane * 16);
    bfrag[0] = *(const v16bf*)(lB[p] + (waveCol * 2 + 0) * 512 + lane * 16);
    bfrag[1] = *(const v16bf*)(lB[p] + (waveCol * 2 + 1) * 512 + lane * 16);

    acc[0][0] = __builtin_amdgcn_wmma_f32_16x16x32_bf16(false, afrag[0], false, bfrag[0], (short)0, acc[0][0], false, false);
    acc[0][1] = __builtin_amdgcn_wmma_f32_16x16x32_bf16(false, afrag[0], false, bfrag[1], (short)0, acc[0][1], false, false);
    acc[1][0] = __builtin_amdgcn_wmma_f32_16x16x32_bf16(false, afrag[1], false, bfrag[0], (short)0, acc[1][0], false, false);
    acc[1][1] = __builtin_amdgcn_wmma_f32_16x16x32_bf16(false, afrag[1], false, bfrag[1], (short)0, acc[1][1], false, false);

    wait_async0();      // next B tile resident in LDS
    __syncthreads();    // all A stores/frag reads for this phase done
  }

  // ---- epilogue: C layout lane<16: n=l16, M=v ; lane>=16: n=l16, M=v+8 ----
#pragma unroll
  for (int i = 0; i < 2; i++) {
#pragma unroll
    for (int j = 0; j < 2; j++) {
      int gc = blockN + waveCol * 32 + j * 16 + l16;
#pragma unroll
      for (int v = 0; v < 8; v++) {
        int grr = blockM + waveRow * 32 + i * 16 + v + half * 8;
        if (grr < M) {
          float val = acc[i][j][v];
          if (bias) val += bias[gc];
          if (Cin)  val += Cin[(long)grr * ldc + gc];
          if (act == 1)      val = fmaxf(val, 0.0f);
          else if (act == 2) val = tanhf(val);
          C[(long)grr * ldc + gc] = val;
        }
      }
    }
  }
}

// ---------------- LayerNorm (+optional vec add, residual, ELU/ReLU) ----------------
__global__ __launch_bounds__(256)
void ln_act(const float* __restrict__ in, const float* __restrict__ addvec,
            const float* __restrict__ resid,
            const float* __restrict__ g, const float* __restrict__ b,
            float* __restrict__ out, int rows, int cols, int act) {
  int wpb = blockDim.x >> 5;
  int row = blockIdx.x * wpb + (threadIdx.x >> 5);
  int lane = threadIdx.x & 31;
  if (row >= rows) return;
  int per = cols >> 5;
  float vals[8];
  float sum = 0.0f;
  const float* rp = in + (long)row * cols;
  for (int i = 0; i < per; i++) {
    int c = lane + i * 32;
    float v = rp[c];
    if (addvec) v += addvec[c];
    if (resid)  v += resid[(long)row * cols + c];
    vals[i] = v; sum += v;
  }
  for (int o = 16; o >= 1; o >>= 1) sum += __shfl_xor(sum, o, 32);
  float mu = sum / (float)cols;
  float vs = 0.0f;
  for (int i = 0; i < per; i++) { float d = vals[i] - mu; vs += d * d; }
  for (int o = 16; o >= 1; o >>= 1) vs += __shfl_xor(vs, o, 32);
  float inv = rsqrtf(vs / (float)cols + 1e-5f);
  for (int i = 0; i < per; i++) {
    int c = lane + i * 32;
    float v = (vals[i] - mu) * inv * g[c] + b[c];
    if (act == 1)      v = (v > 0.0f) ? v : (expf(v) - 1.0f);  // ELU
    else if (act == 2) v = fmaxf(v, 0.0f);                      // ReLU
    out[(long)row * cols + c] = v;
  }
}

// ---------------- LSTM cell (gate order i,f,g,o) ----------------
__global__ void lstm_cell(const float* __restrict__ gates, float* __restrict__ h,
                          float* __restrict__ c, int N, int H) {
  int idx = blockIdx.x * blockDim.x + threadIdx.x;
  if (idx >= N * H) return;
  int n = idx / H, j = idx - n * H;
  const float* gp = gates + (long)n * 4 * H;
  float ig = gp[j], fg = gp[H + j], gg = gp[2 * H + j], og = gp[3 * H + j];
  float si = 1.0f / (1.0f + expf(-ig));
  float sf = 1.0f / (1.0f + expf(-fg));
  float so = 1.0f / (1.0f + expf(-og));
  float cn = sf * c[idx] + si * tanhf(gg);
  c[idx] = cn;
  h[idx] = so * tanhf(cn);
}

// ---------------- fills ----------------
__global__ void fill_f32(float* p, float v, long n) {
  long i = (long)blockIdx.x * blockDim.x + threadIdx.x; if (i < n) p[i] = v;
}
__global__ void fill_u32(unsigned* p, unsigned v, long n) {
  long i = (long)blockIdx.x * blockDim.x + threadIdx.x; if (i < n) p[i] = v;
}

// ---------------- GATv2 edge kernels (H=4, D=64, Hd=256) ----------------
__global__ __launch_bounds__(256)
void gat_edge_score(const float* __restrict__ xl, const float* __restrict__ xr,
                    const float* __restrict__ att, const int* __restrict__ ei,
                    int E, int Etot, float* __restrict__ sc, unsigned* __restrict__ m) {
  int wpb = blockDim.x >> 5;
  int e = blockIdx.x * wpb + (threadIdx.x >> 5);
  int lane = threadIdx.x & 31;
  if (e >= Etot) return;
  int s, dn; edge_sd(ei, E, e, s, dn);
  const float* xls = xl + (long)s * 256;
  const float* xrd = xr + (long)dn * 256;
  for (int h = 0; h < 4; h++) {
    float acc = 0.0f;
    for (int q = 0; q < 2; q++) {
      int dd = lane + q * 32;
      float v = xls[h * 64 + dd] + xrd[h * 64 + dd];
      v = (v > 0.0f) ? v : 0.2f * v;                 // leaky_relu 0.2
      acc += v * att[h * 64 + dd];
    }
    for (int o = 16; o >= 1; o >>= 1) acc += __shfl_xor(acc, o, 32);
    if (lane == 0) {
      sc[(long)e * 4 + h] = acc;
      atomicMax(&m[dn * 4 + h], fenc(acc));
    }
  }
}

__global__ void gat_edge_exp(float* __restrict__ sc, const int* __restrict__ ei,
                             int E, int Etot, const unsigned* __restrict__ m,
                             float* __restrict__ z) {
  long idx = (long)blockIdx.x * blockDim.x + threadIdx.x;
  if (idx >= (long)Etot * 4) return;
  int e = (int)(idx >> 2), h = (int)(idx & 3);
  int s, dn; edge_sd(ei, E, e, s, dn);
  unsigned mu_u = m[dn * 4 + h];
  float mu = (mu_u == ENC_NEGINF) ? 0.0f : fdec(mu_u);
  float ev = expf(sc[idx] - mu);
  sc[idx] = ev;
  atomicAdd(&z[dn * 4 + h], ev);
}

__global__ void gat_edge_accum(const float* __restrict__ sc, const float* __restrict__ xl,
                               const int* __restrict__ ei, int E, int Etot,
                               const float* __restrict__ z, float* __restrict__ out) {
  long idx = (long)blockIdx.x * blockDim.x + threadIdx.x;
  if (idx >= (long)Etot * 256) return;
  int e = (int)(idx >> 8); int q = (int)(idx & 255); int h = q >> 6;
  int s, dn; edge_sd(ei, E, e, s, dn);
  float alpha = sc[(long)e * 4 + h] / (z[dn * 4 + h] + 1e-16f);
  atomicAdd(&out[(long)dn * 256 + q], alpha * xl[(long)s * 256 + q]);
}

// ---------------- attention pooling ----------------
__global__ __launch_bounds__(256)
void pool_score(const float* __restrict__ t1, const float* __restrict__ Wa2,
                const float* __restrict__ ba2, const int* __restrict__ batch,
                float* __restrict__ scn, unsigned* __restrict__ msec, int N) {
  int wpb = blockDim.x >> 5;
  int n = blockIdx.x * wpb + (threadIdx.x >> 5);
  int lane = threadIdx.x & 31;
  if (n >= N) return;
  float acc = 0.0f;
  for (int q = 0; q < 4; q++) { int c = lane + q * 32; acc += t1[(long)n * 128 + c] * Wa2[c]; }
  for (int o = 16; o >= 1; o >>= 1) acc += __shfl_xor(acc, o, 32);
  if (lane == 0) {
    acc += ba2[0];
    scn[n] = acc;
    atomicMax(&msec[batch[n]], fenc(acc));
  }
}

__global__ void pool_exp(const float* __restrict__ scn, const int* __restrict__ batch,
                         const unsigned* __restrict__ msec, float* __restrict__ zsec,
                         float* __restrict__ cnt, float* __restrict__ es, int N) {
  int n = blockIdx.x * blockDim.x + threadIdx.x;
  if (n >= N) return;
  int s = batch[n];
  unsigned mu_u = msec[s];
  float mu = (mu_u == ENC_NEGINF) ? 0.0f : fdec(mu_u);
  float e = expf(scn[n] - mu);
  es[n] = e;
  atomicAdd(&zsec[s], e);
  atomicAdd(&cnt[s], 1.0f);
}

__global__ void pool_accum(const float* __restrict__ es, const float* __restrict__ zsec,
                           const int* __restrict__ batch, const float* __restrict__ x,
                           float* __restrict__ se, int N) {
  long idx = (long)blockIdx.x * blockDim.x + threadIdx.x;
  if (idx >= (long)N * 256) return;
  int n = (int)(idx >> 8);
  int s = batch[n];
  float w = es[n] / (zsec[s] + 1e-16f);
  atomicAdd(&se[(long)s * 256 + (idx & 255)], w * x[idx]);
}

__global__ void pool_final(float* __restrict__ se, const float* __restrict__ cnt, int S) {
  int idx = blockIdx.x * blockDim.x + threadIdx.x;
  if (idx >= S * 256) return;
  se[idx] /= fmaxf(cnt[idx >> 8], 1.0f);
}

// ---------------- fusion gather ----------------
__global__ void gather_comb(const float* __restrict__ x, const float* __restrict__ se,
                            const int* __restrict__ batch, float* __restrict__ comb, int N) {
  long idx = (long)blockIdx.x * blockDim.x + threadIdx.x;
  if (idx >= (long)N * 512) return;
  int n = (int)(idx >> 9); int c = (int)(idx & 511);
  comb[idx] = (c < 256) ? x[(long)n * 256 + c] : se[(long)batch[n] * 256 + (c - 256)];
}

// ---------------- final heads ----------------
__global__ __launch_bounds__(256)
void final_heads(const float* __restrict__ h64, const float* __restrict__ hm,
                 const float* __restrict__ hk,
                 const float* __restrict__ Wr3, const float* __restrict__ br3,
                 const float* __restrict__ Wc2, const float* __restrict__ bc2,
                 const float* __restrict__ Wk2, const float* __restrict__ bk2,
                 float* __restrict__ outp, int N) {
  int wpb = blockDim.x >> 5;
  int n = blockIdx.x * wpb + (threadIdx.x >> 5);
  int lane = threadIdx.x & 31;
  if (n >= N) return;
  float a = h64[(long)n * 64 + lane] * Wr3[lane]
          + h64[(long)n * 64 + 32 + lane] * Wr3[32 + lane];
  float m0 = 0.0f, m1 = 0.0f, r = 0.0f;
  for (int q = 0; q < 4; q++) {
    int c = lane + q * 32;
    float hv = hm[(long)n * 128 + c];
    m0 += hv * Wc2[c * 2 + 0];
    m1 += hv * Wc2[c * 2 + 1];
    r  += hk[(long)n * 128 + c] * Wk2[c];
  }
  for (int o = 16; o >= 1; o >>= 1) {
    a  += __shfl_xor(a, o, 32);
    m0 += __shfl_xor(m0, o, 32);
    m1 += __shfl_xor(m1, o, 32);
    r  += __shfl_xor(r, o, 32);
  }
  if (lane == 0) {
    outp[n]              = tanhf(a + br3[0]) * 0.1f;
    outp[N + 2 * n]      = m0 + bc2[0];
    outp[N + 2 * n + 1]  = m1 + bc2[1];
    outp[3 * N + n]      = 1.0f / (1.0f + expf(-(r + bk2[0])));
  }
}

// ---------------- host-side helpers ----------------
static inline void gemm(hipStream_t st, const float* A, int lda, const bf16_t* Bp,
                        const float* bias, const float* Cin,
                        float* C, int ldc, int M, int Ncol, int K, int act) {
  dim3 grid((Ncol + 63) / 64, (M + 63) / 64);
  gemm_wmma<<<grid, dim3(128), 0, st>>>(A, lda, Bp, bias, Cin, C, ldc, M, Ncol, K, act);
}

extern "C" void kernel_launch(void* const* d_in, const int* in_sizes, int n_in,
                              void* d_out, int out_size, void* d_ws, size_t ws_size,
                              hipStream_t stream) {
  // ---- inputs (setup_inputs dict order) ----
  const float* x_stock   = (const float*)d_in[0];
  const int*   ei_stock  = (const int*)d_in[1];
  const int*   batch     = (const int*)d_in[2];
  const int*   ei_sector = (const int*)d_in[4];   // d_in[3] x_sector unused by forward
  const float* ln_feat_g = (const float*)d_in[5];
  const float* ln_feat_b = (const float*)d_in[6];
  const float* w_ih0 = (const float*)d_in[7];  const float* w_hh0 = (const float*)d_in[8];
  const float* b_ih0 = (const float*)d_in[9];  const float* b_hh0 = (const float*)d_in[10];
  const float* w_ih1 = (const float*)d_in[11]; const float* w_hh1 = (const float*)d_in[12];
  const float* b_ih1 = (const float*)d_in[13]; const float* b_hh1 = (const float*)d_in[14];
  const float* ln_t_g = (const float*)d_in[15]; const float* ln_t_b = (const float*)d_in[16];
  const float* W_in = (const float*)d_in[17];  const float* b_in = (const float*)d_in[18];
  const float* stock_Wl = (const float*)d_in[19]; const float* stock_bl = (const float*)d_in[20];
  const float* stock_Wr = (const float*)d_in[21]; const float* stock_br = (const float*)d_in[22];
  const float* stock_att = (const float*)d_in[23]; const float* stock_bias = (const float*)d_in[24];
  const float* stock_g = (const float*)d_in[25]; const float* stock_b = (const float*)d_in[26];
  const float* sector_Wl = (const float*)d_in[27]; const float* sector_bl = (const float*)d_in[28];
  const float* sector_Wr = (const float*)d_in[29]; const float* sector_br = (const float*)d_in[30];
  const float* sector_att = (const float*)d_in[31]; const float* sector_bias = (const float*)d_in[32];
  const float* sector_g = (const float*)d_in[33]; const float* sector_b = (const float*)d_in[34];
  const float* Wa1 = (const float*)d_in[35]; const float* ba1 = (const float*)d_in[36];
  const float* Wa2 = (const float*)d_in[37]; const float* ba2 = (const float*)d_in[38];
  const float* Wf = (const float*)d_in[39];  const float* bfv = (const float*)d_in[40];
  const float* gf = (const float*)d_in[41];  const float* bef = (const float*)d_in[42];
  const float* W_regf = (const float*)d_in[43]; const float* b_regf = (const float*)d_in[44];
  const float* g_regf = (const float*)d_in[45]; const float* be_regf = (const float*)d_in[46];
  const float* W_clff = (const float*)d_in[47]; const float* b_clff = (const float*)d_in[48];
  const float* g_clff = (const float*)d_in[49]; const float* be_clff = (const float*)d_in[50];
  const float* W_rnkf = (const float*)d_in[51]; const float* b_rnkf = (const float*)d_in[52];
  const float* g_rnkf = (const float*)d_in[53]; const float* be_rnkf = (const float*)d_in[54];
  const float* Wr1 = (const float*)d_in[55]; const float* br1 = (const float*)d_in[56];
  const float* Wr2 = (const float*)d_in[57]; const float* br2 = (const float*)d_in[58];
  const float* Wr3 = (const float*)d_in[59]; const float* br3 = (const float*)d_in[60];
  const float* Wc1 = (const float*)d_in[61]; const float* bc1 = (const float*)d_in[62];
  const float* Wc2 = (const float*)d_in[63]; const float* bc2 = (const float*)d_in[64];
  const float* Wk1 = (const float*)d_in[65]; const float* bk1 = (const float*)d_in[66];
  const float* Wk2 = (const float*)d_in[67]; const float* bk2 = (const float*)d_in[68];

  const int N = 20000, T = 20, F = 64, Hd = 256, E = 640000, S = 64, ES = 1024, L = 2;

  // ---- workspace layout: fp32 arena then bf16 arena ----
  float* ws = (float*)d_ws;
  size_t off = 0;
  auto alloc = [&](size_t n) { size_t o = off; off += (n + 255) & ~(size_t)255; return ws + o; };
  float* lnx    = alloc((size_t)N * T * F);
  float* gates  = alloc((size_t)N * 4 * Hd);
  float* h0 = alloc((size_t)N * Hd);
  float* c0 = alloc((size_t)N * Hd);
  float* h1 = alloc((size_t)N * Hd);
  float* c1 = alloc((size_t)N * Hd);
  float* xbuf = alloc((size_t)N * Hd);
  float* xlb  = alloc((size_t)N * Hd);
  float* xrb  = alloc((size_t)N * Hd);
  float* aggb = alloc((size_t)N * Hd);
  float* esc  = alloc((size_t)(E + N) * 4);
  unsigned* mbuf = (unsigned*)alloc((size_t)N * 4);
  float* zbuf = alloc((size_t)N * 4);
  float* t1buf = alloc((size_t)N * 128);
  float* scn = alloc((size_t)N);
  float* esn = alloc((size_t)N);
  unsigned* msec = (unsigned*)alloc(64);
  float* zsec = alloc(64);
  float* cnt  = alloc(64);
  float* se   = alloc((size_t)S * Hd);
  float* xls  = alloc((size_t)S * Hd);
  float* xrs  = alloc((size_t)S * Hd);
  float* aggs = alloc((size_t)S * Hd);
  float* escs = alloc((size_t)(ES + S) * 4);
  unsigned* ms = (unsigned*)alloc((size_t)S * 4);
  float* zs = alloc((size_t)S * 4);
  float* comb = alloc((size_t)N * 2 * Hd);
  float* tmp256 = alloc((size_t)N * Hd);
  float* feat = alloc((size_t)N * Hd);
  float* h64  = alloc((size_t)N * 64);
  float* hmov = alloc((size_t)N * 128);
  float* hrnk = alloc((size_t)N * 128);
  (void)ws_size; (void)n_in; (void)in_sizes; (void)out_size;

  // bf16 arena for packed weights
  bf16_t* barena = (bf16_t*)(ws + off);
  size_t boff = 0;
  auto balloc = [&](size_t n) { bf16_t* p = barena + boff; boff += (n + 1023) & ~(size_t)1023; return p; };
  auto pack = [&](const float* W, int ldw, int trans, int K, int Ncol) {
    bf16_t* dst = balloc((size_t)K * Ncol);
    pack_b<<<(int)(((long)K * Ncol + 255) / 256), 256, 0, stream>>>(W, ldw, trans, dst, K, Ncol);
    return dst;
  };

  bf16_t* pb_wih0 = pack(w_ih0, F, 1, F, 4 * Hd);
  bf16_t* pb_whh0 = pack(w_hh0, Hd, 1, Hd, 4 * Hd);
  bf16_t* pb_wih1 = pack(w_ih1, Hd, 1, Hd, 4 * Hd);
  bf16_t* pb_whh1 = pack(w_hh1, Hd, 1, Hd, 4 * Hd);
  bf16_t* pb_Win  = pack(W_in, Hd, 0, Hd, Hd);
  bf16_t* pb_sWl[2]; bf16_t* pb_sWr[2]; bf16_t* pb_cWl[2]; bf16_t* pb_cWr[2];
  for (int l = 0; l < L; l++) {
    pb_sWl[l] = pack(stock_Wl + (size_t)l * Hd * Hd, Hd, 0, Hd, Hd);
    pb_sWr[l] = pack(stock_Wr + (size_t)l * Hd * Hd, Hd, 0, Hd, Hd);
    pb_cWl[l] = pack(sector_Wl + (size_t)l * Hd * Hd, Hd, 0, Hd, Hd);
    pb_cWr[l] = pack(sector_Wr + (size_t)l * Hd * Hd, Hd, 0, Hd, Hd);
  }
  bf16_t* pb_Wa1 = pack(Wa1, 128, 0, Hd, 128);
  bf16_t* pb_Wf  = pack(Wf, Hd, 0, 2 * Hd, Hd);
  bf16_t* pb_Wregf = pack(W_regf, Hd, 0, Hd, Hd);
  bf16_t* pb_Wclff = pack(W_clff, Hd, 0, Hd, Hd);
  bf16_t* pb_Wrnkf = pack(W_rnkf, Hd, 0, Hd, Hd);
  bf16_t* pb_Wr1 = pack(Wr1, 128, 0, Hd, 128);
  bf16_t* pb_Wr2 = pack(Wr2, 64, 0, 128, 64);
  bf16_t* pb_Wc1 = pack(Wc1, 128, 0, Hd, 128);
  bf16_t* pb_Wk1 = pack(Wk1, 128, 0, Hd, 128);

  auto fillf = [&](float* p, float v, long n) {
    fill_f32<<<(int)((n + 255) / 256), 256, 0, stream>>>(p, v, n);
  };
  auto fillu = [&](unsigned* p, unsigned v, long n) {
    fill_u32<<<(int)((n + 255) / 256), 256, 0, stream>>>(p, v, n);
  };

  // ---- 1. feature LayerNorm over [N*T, F] ----
  ln_act<<<(N * T + 7) / 8, 256, 0, stream>>>(x_stock, nullptr, nullptr,
                                              ln_feat_g, ln_feat_b, lnx, N * T, F, 0);

  // ---- 2. two-layer LSTM, interleaved per timestep ----
  fillf(h0, 0.0f, (long)N * Hd); fillf(c0, 0.0f, (long)N * Hd);
  fillf(h1, 0.0f, (long)N * Hd); fillf(c1, 0.0f, (long)N * Hd);
  for (int t = 0; t < T; t++) {
    gemm(stream, lnx + t * F, T * F, pb_wih0, b_ih0, nullptr, gates, 4 * Hd, N, 4 * Hd, F, 0);
    gemm(stream, h0, Hd, pb_whh0, b_hh0, gates, gates, 4 * Hd, N, 4 * Hd, Hd, 0);
    lstm_cell<<<(N * Hd + 255) / 256, 256, 0, stream>>>(gates, h0, c0, N, Hd);
    gemm(stream, h0, Hd, pb_wih1, b_ih1, nullptr, gates, 4 * Hd, N, 4 * Hd, Hd, 0);
    gemm(stream, h1, Hd, pb_whh1, b_hh1, gates, gates, 4 * Hd, N, 4 * Hd, Hd, 0);
    lstm_cell<<<(N * Hd + 255) / 256, 256, 0, stream>>>(gates, h1, c1, N, Hd);
  }

  // ---- 3. temporal LN + input projection ----
  ln_act<<<(N + 7) / 8, 256, 0, stream>>>(h1, nullptr, nullptr, ln_t_g, ln_t_b, tmp256, N, Hd, 0);
  gemm(stream, tmp256, Hd, pb_Win, b_in, nullptr, xbuf, Hd, N, Hd, Hd, 0);

  // ---- 4. stock GATv2 stack ----
  auto gat_layer = [&](float* xio, int Nn, const int* ei, int Ee,
                       const bf16_t* pWl, const float* bl, const bf16_t* pWr, const float* br,
                       const float* att, const float* bias_, const float* g_, const float* b_,
                       float* xl_, float* xr_, float* agg_, float* sc_, unsigned* m_, float* z_) {
    int Etot = Ee + Nn;
    gemm(stream, xio, 256, pWl, bl, nullptr, xl_, 256, Nn, 256, 256, 0);
    gemm(stream, xio, 256, pWr, br, nullptr, xr_, 256, Nn, 256, 256, 0);
    fillu(m_, ENC_NEGINF, (long)Nn * 4);
    fillf(z_, 0.0f, (long)Nn * 4);
    fillf(agg_, 0.0f, (long)Nn * 256);
    gat_edge_score<<<(Etot + 7) / 8, 256, 0, stream>>>(xl_, xr_, att, ei, Ee, Etot, sc_, m_);
    gat_edge_exp<<<(int)(((long)Etot * 4 + 255) / 256), 256, 0, stream>>>(sc_, ei, Ee, Etot, m_, z_);
    gat_edge_accum<<<(int)(((long)Etot * 256 + 255) / 256), 256, 0, stream>>>(sc_, xl_, ei, Ee, Etot, z_, agg_);
    ln_act<<<(Nn + 7) / 8, 256, 0, stream>>>(agg_, bias_, xio, g_, b_, xio, Nn, 256, 1);
  };
  for (int l = 0; l < L; l++)
    gat_layer(xbuf, N, ei_stock, E,
              pb_sWl[l], stock_bl + l * Hd, pb_sWr[l], stock_br + l * Hd,
              stock_att + l * Hd, stock_bias + l * Hd, stock_g + l * Hd, stock_b + l * Hd,
              xlb, xrb, aggb, esc, mbuf, zbuf);

  // ---- 5. attention pooling per sector ----
  gemm(stream, xbuf, 256, pb_Wa1, ba1, nullptr, t1buf, 128, N, 128, 256, 2 /*tanh*/);
  fillu(msec, ENC_NEGINF, 64);
  fillf(zsec, 0.0f, 64); fillf(cnt, 0.0f, 64); fillf(se, 0.0f, (long)S * Hd);
  pool_score<<<(N + 7) / 8, 256, 0, stream>>>(t1buf, Wa2, ba2, batch, scn, msec, N);
  pool_exp<<<(N + 255) / 256, 256, 0, stream>>>(scn, batch, msec, zsec, cnt, esn, N);
  pool_accum<<<(int)(((long)N * 256 + 255) / 256), 256, 0, stream>>>(esn, zsec, batch, xbuf, se, N);
  pool_final<<<(S * 256 + 255) / 256, 256, 0, stream>>>(se, cnt, S);

  // ---- 6. sector GATv2 stack ----
  for (int l = 0; l < L; l++)
    gat_layer(se, S, ei_sector, ES,
              pb_cWl[l], sector_bl + l * Hd, pb_cWr[l], sector_br + l * Hd,
              sector_att + l * Hd, sector_bias + l * Hd, sector_g + l * Hd, sector_b + l * Hd,
              xls, xrs, aggs, escs, ms, zs);

  // ---- 7. fusion -> c (written straight into d_out) ----
  gather_comb<<<(int)(((long)N * 512 + 255) / 256), 256, 0, stream>>>(xbuf, se, batch, comb, N);
  gemm(stream, comb, 512, pb_Wf, bfv, nullptr, tmp256, 256, N, 256, 512, 0);
  float* cptr = (float*)d_out + (size_t)4 * N;  // layout: ret[N] | mov[N*2] | rnk[N] | c[N*256]
  ln_act<<<(N + 7) / 8, 256, 0, stream>>>(tmp256, nullptr, nullptr, gf, bef, cptr, N, 256, 1);

  // ---- 8. heads ----
  gemm(stream, cptr, 256, pb_Wregf, b_regf, nullptr, tmp256, 256, N, 256, 256, 0);
  ln_act<<<(N + 7) / 8, 256, 0, stream>>>(tmp256, nullptr, nullptr, g_regf, be_regf, feat, N, 256, 2);
  gemm(stream, feat, 256, pb_Wr1, br1, nullptr, t1buf, 128, N, 128, 256, 1 /*relu*/);
  gemm(stream, t1buf, 128, pb_Wr2, br2, nullptr, h64, 64, N, 64, 128, 1 /*relu*/);

  gemm(stream, cptr, 256, pb_Wclff, b_clff, nullptr, tmp256, 256, N, 256, 256, 0);
  ln_act<<<(N + 7) / 8, 256, 0, stream>>>(tmp256, nullptr, nullptr, g_clff, be_clff, feat, N, 256, 2);
  gemm(stream, feat, 256, pb_Wc1, bc1, nullptr, hmov, 128, N, 128, 256, 1 /*relu*/);

  gemm(stream, cptr, 256, pb_Wrnkf, b_rnkf, nullptr, tmp256, 256, N, 256, 256, 0);
  ln_act<<<(N + 7) / 8, 256, 0, stream>>>(tmp256, nullptr, nullptr, g_rnkf, be_rnkf, feat, N, 256, 2);
  gemm(stream, feat, 256, pb_Wk1, bk1, nullptr, hrnk, 128, N, 128, 256, 1 /*relu*/);

  final_heads<<<(N + 7) / 8, 256, 0, stream>>>(h64, hmov, hrnk, Wr3, br3, Wc2, bc2, Wk2, bk2,
                                               (float*)d_out, N);
}
